// EdgeConvFeatures_65764539236699
// MI455X (gfx1250) — matrix operand order repaired and checked by reading, not compiled, not run on hardware
//
#include <hip/hip_runtime.h>

#define KNN 10
#define NPTS 2048
#define NB 16

typedef __attribute__((ext_vector_type(16))) _Float16 v16h;
typedef __attribute__((ext_vector_type(8)))  float    v8f;
typedef __attribute__((ext_vector_type(4)))  float    f32x4;
typedef __attribute__((ext_vector_type(4)))  unsigned int u32x4;

union FragH { v16h v; f32x4 q[2]; };

// ---------------- kNN: one wave32 per point ----------------
template<int C>
__global__ __launch_bounds__(128)
void knn_kernel(const float* __restrict__ h, int* __restrict__ out, int N)
{
    const int wave = threadIdx.x >> 5, lane = threadIdx.x & 31;
    const int p = blockIdx.x * 4 + wave;      // grid sized exactly: p < B*N
    const int b = p / N, n = p - b * N;
    const float* xb = h + (size_t)b * N * C;

    float xi[C];
#pragma unroll
    for (int c = 0; c < C; ++c) xi[c] = xb[(size_t)n * C + c];

    float dk[KNN]; int ik[KNN];
#pragma unroll
    for (int j = 0; j < KNN; ++j) { dk[j] = 3.0e38f; ik[j] = 0; }

    // d(n,m) ranking key: |x_m|^2 - 2 x_n . x_m   (|x_n|^2 is constant)
    for (int m = lane; m < N; m += 32) {
        const float* xm = xb + (size_t)m * C;
        float dot = 0.f, sq = 0.f;
#pragma unroll
        for (int c = 0; c < C; ++c) { float v = xm[c]; dot += xi[c] * v; sq += v * v; }
        float s = sq - 2.0f * dot;
        if (s < dk[KNN - 1]) {
            dk[KNN - 1] = s; ik[KNN - 1] = m;
#pragma unroll
            for (int j = KNN - 1; j > 0; --j) {
                if (dk[j] < dk[j - 1]) {
                    float td = dk[j]; dk[j] = dk[j - 1]; dk[j - 1] = td;
                    int   ti = ik[j]; ik[j] = ik[j - 1]; ik[j - 1] = ti;
                }
            }
        }
    }

    // wave-wide 10-way merge: repeated argmin over per-lane sorted heads
    int cur = 0;
    for (int t = 0; t < KNN; ++t) {
        float v = 3.0e38f; int vi = 0;
#pragma unroll
        for (int j = 0; j < KNN; ++j) if (cur == j) { v = dk[j]; vi = ik[j]; }
        int wl = lane;
#pragma unroll
        for (int off = 16; off > 0; off >>= 1) {
            float ov = __shfl_xor(v, off, 32);
            int   oi = __shfl_xor(vi, off, 32);
            int   ow = __shfl_xor(wl, off, 32);
            if (ov < v || (ov == v && ow < wl)) { v = ov; vi = oi; wl = ow; }
        }
        if (lane == wl) cur++;
        if (lane == 0) out[(size_t)p * KNN + t] = vi;
    }
}

// ---------------- Edge-feature gather: e = [x_i, x_j - x_i] (f16, K-padded) ----------------
__global__ void gather_kernel(const float* __restrict__ h, const int* __restrict__ kn,
                              _Float16* __restrict__ E, int N, int C, int kpad, int total)
{
    int r = blockIdx.x * blockDim.x + threadIdx.x;
    if (r >= total) return;
    int p = r / KNN;
    int b = p / N, n = p - b * N;
    const float* xb = h + (size_t)b * N * C;
    int j = kn[r];
    const float* xi = xb + (size_t)n * C;
    const float* xj = xb + (size_t)j * C;
    _Float16* e = E + (size_t)r * kpad;
    for (int c = 0; c < C; ++c) {
        float a = xi[c];
        e[c]     = (_Float16)a;
        e[C + c] = (_Float16)(xj[c] - a);
    }
    for (int c = 2 * C; c < kpad; ++c) e[c] = (_Float16)0.f;
}

// ---------------- WMMA GEMM + bias + ReLU + BN partial stats ----------------
// X: M x KPAD f16 (row-major, zero-padded K). W: kdim x cout f32. Y: M x ldy f16.
// grid = (M/64, cout/16), block = 128 (4 waves, one 16-row tile each).
template<int KPAD>
__global__ __launch_bounds__(128)
void gemm_bn_kernel(const _Float16* __restrict__ X, const float* __restrict__ W,
                    const float* __restrict__ bias, _Float16* __restrict__ Y,
                    float* __restrict__ ssum, float* __restrict__ ssq,
                    int kdim, int cout, int ldy)
{
    __shared__ _Float16 ldsA[4][16 * KPAD];   // per-wave A tile (row-major)
    __shared__ _Float16 ldsB[16 * KPAD];      // B^T: [n][k], K contiguous
    __shared__ float redS[16], redQ[16];

    const int tid = threadIdx.x;
    const int wave = tid >> 5, lane = tid & 31;
    const int n0 = blockIdx.y * 16;
    const int m0 = (blockIdx.x * 4 + wave) * 16;

    // stage W (f32 -> f16, transposed, K zero-padded)
    for (int i = tid; i < 16 * KPAD; i += 128) {
        int n = i / KPAD, k = i - n * KPAD;
        float wv = (k < kdim) ? W[k * cout + n0 + n] : 0.0f;
        ldsB[i] = (_Float16)wv;
    }
    // stage A: 16 rows x KPAD halves is one contiguous chunk
    {
        const u32x4* src = (const u32x4*)(X + (size_t)m0 * KPAD);
        u32x4* dst = (u32x4*)(&ldsA[wave][0]);
#pragma unroll
        for (int i = lane; i < 16 * KPAD / 8; i += 32) dst[i] = src[i];
    }
    if (tid < 16) { redS[tid] = 0.f; redQ[tid] = 0.f; }
    __syncthreads();

    const int hi  = lane >> 4;     // lane group 0/1
    const int l16 = lane & 15;
    float bv = bias[n0 + l16];
    v8f c = { bv, bv, bv, bv, bv, bv, bv, bv };

    const _Float16* arow = &ldsA[wave][l16 * KPAD];  // A row M=l16
    const _Float16* bcol = &ldsB[l16 * KPAD];        // B col N=l16
#pragma unroll
    for (int ks = 0; ks < KPAD; ks += 32) {
        FragH a, bf;
        // A 16x32 f16 layout: lanes<16 -> K {0..7,16..23}; lanes>=16 -> K {8..15,24..31}
        a.q[0] = *(const f32x4*)(arow + ks + hi * 8);
        a.q[1] = *(const f32x4*)(arow + ks + 16 + hi * 8);
        // B 32x16 f16 layout: lanes<16 -> K {0..7,16..23}; lanes>=16 -> K {8..15,24..31}
        bf.q[0] = *(const f32x4*)(bcol + ks + hi * 8);
        bf.q[1] = *(const f32x4*)(bcol + ks + 16 + hi * 8);
        c = __builtin_amdgcn_wmma_f32_16x16x32_f16(false, a.v, false, bf.v,
                                                   (short)0, c, false, false);
    }

#pragma unroll
    for (int v = 0; v < 8; ++v) c[v] = fmaxf(c[v], 0.0f);   // ReLU

    // store Y (f16): row M = v + 8*hi, col N = l16
#pragma unroll
    for (int v = 0; v < 8; ++v) {
        int row = v + 8 * hi;
        Y[(size_t)(m0 + row) * ldy + n0 + l16] = (_Float16)c[v];
    }

    // BN partial stats per column
    float s = 0.f, q = 0.f;
#pragma unroll
    for (int v = 0; v < 8; ++v) { s += c[v]; q += c[v] * c[v]; }
    s += __shfl_xor(s, 16, 32);
    q += __shfl_xor(q, 16, 32);
    if (hi == 0) { atomicAdd(&redS[l16], s); atomicAdd(&redQ[l16], q); }
    __syncthreads();
    if (tid < 16) {
        atomicAdd(&ssum[n0 + tid], redS[tid]);
        atomicAdd(&ssq[n0 + tid], redQ[tid]);
    }
}

// ---------------- BN helpers ----------------
__global__ void zstats_kernel(float* s) { s[threadIdx.x] = 0.f; }   // 128 threads: sum+sumsq

__global__ void stats_fin_kernel(const float* __restrict__ ssum, const float* __restrict__ ssq,
                                 float* __restrict__ scl, float* __restrict__ shf,
                                 const float* __restrict__ gamma, const float* __restrict__ beta,
                                 int cout, float Mf)
{
    int c = threadIdx.x;
    if (c < cout) {
        float mu  = ssum[c] / Mf;
        float var = ssq[c] / Mf - mu * mu;
        float s   = gamma[c] * rsqrtf(var + 1e-5f);
        scl[c] = s;
        shf[c] = beta[c] - mu * s;
    }
}

__global__ void norm_kernel(_Float16* __restrict__ Y, const float* __restrict__ scl,
                            const float* __restrict__ shf, long total, int cout)
{
    long i = (long)blockIdx.x * blockDim.x + threadIdx.x;
    if (i < total) {
        int c = (int)(i % cout);
        float v = (float)Y[i];
        Y[i] = (_Float16)(v * scl[c] + shf[c]);
    }
}

// normalize last MLP layer + max over K neighbors -> h (B*N, 32) f32
__global__ void maxk_kernel(const _Float16* __restrict__ Y, const float* __restrict__ scl,
                            const float* __restrict__ shf, float* __restrict__ hout, int total)
{
    int i = blockIdx.x * blockDim.x + threadIdx.x;
    if (i >= total) return;
    int c = i & 31, p = i >> 5;
    float s = scl[c], f = shf[c];
    float m = -3.0e38f;
#pragma unroll
    for (int k = 0; k < KNN; ++k) {
        float v = (float)Y[((size_t)p * KNN + k) * 32 + c] * s + f;
        m = fmaxf(m, v);
    }
    hout[i] = m;
}

// global max over N points: one block per (b,c)
__global__ void gmax_kernel(const float* __restrict__ h, float* __restrict__ g)
{
    __shared__ float red[256];
    int o = blockIdx.x, b = o >> 5, c = o & 31;
    float m = -3.0e38f;
    for (int n = threadIdx.x; n < NPTS; n += 256)
        m = fmaxf(m, h[((size_t)b * NPTS + n) * 32 + c]);
    red[threadIdx.x] = m; __syncthreads();
    for (int s = 128; s > 0; s >>= 1) {
        if (threadIdx.x < s) red[threadIdx.x] = fmaxf(red[threadIdx.x], red[threadIdx.x + s]);
        __syncthreads();
    }
    if (threadIdx.x == 0) g[o] = red[0];
}

// final linear: out[b,o] = g[b,:] @ W[:,o] + bias[o]   (16x512, f32)
__global__ void linear_kernel(const float* __restrict__ g, const float* __restrict__ W,
                              const float* __restrict__ bias, float* __restrict__ out)
{
    int t = blockIdx.x * blockDim.x + threadIdx.x;
    if (t >= NB * 512) return;
    int b = t >> 9, o = t & 511;
    float acc = bias[o];
#pragma unroll 8
    for (int c = 0; c < 32; ++c) acc += g[b * 32 + c] * W[c * 512 + o];
    out[t] = acc;
}

extern "C" void kernel_launch(void* const* d_in, const int* in_sizes, int n_in,
                              void* d_out, int out_size, void* d_ws, size_t ws_size,
                              hipStream_t stream)
{
    const size_t Mrows = (size_t)NB * NPTS * KNN;   // 327680 edge rows
    const int N = NPTS;

    // Input pointer layout: handle both "positions first, insertion order"
    // (W,b,gamma,beta) and "params first, tree-sorted" (W,b,beta,gamma).
    bool insertion = (in_sizes[0] == NB * NPTS * 3);
    int pb = insertion ? 1 : 0;
    const float* pos = (const float*)d_in[insertion ? 0 : (n_in - 1)];
    const int iGamma = insertion ? 2 : 3;
    const int iBeta  = insertion ? 3 : 2;
    auto P = [&](int conv, int layer, int field) -> const float* {
        return (const float*)d_in[pb + conv * 12 + layer * 4 + field];
    };
    const float* linW = (const float*)d_in[pb + 36];
    const float* linB = (const float*)d_in[pb + 37];

    // Workspace carve-up (~94 MB)
    _Float16* pingA = (_Float16*)d_ws;                      // Mrows x 64 f16
    _Float16* pingB = pingA + Mrows * 64;                   // Mrows x 64 f16
    float* hA  = (float*)(pingB + Mrows * 64);              // B*N*32 f32
    float* hB  = hA + (size_t)NB * NPTS * 32;               // B*N*32 f32
    int*   kidx = (int*)(hB + (size_t)NB * NPTS * 32);      // Mrows int
    float* ssum = (float*)(kidx + Mrows);                   // 64
    float* ssq  = ssum + 64;                                // 64
    float* scl  = ssq + 64;                                 // 64
    float* shf  = scl + 64;                                 // 64
    float* gbuf = shf + 64;                                 // B*32

    const float Mf = (float)Mrows;
    const int knnBlocks = NB * NPTS / 4;

    auto run_conv = [&](const float* hin, int cin, int conv, float* hout) {
        const int cin2 = 2 * cin;
        const int kpad = (cin2 <= 32) ? 32 : 64;

        if (cin == 3) knn_kernel<3><<<knnBlocks, 128, 0, stream>>>(hin, kidx, N);
        else          knn_kernel<32><<<knnBlocks, 128, 0, stream>>>(hin, kidx, N);

        gather_kernel<<<(int)((Mrows + 255) / 256), 256, 0, stream>>>(
            hin, kidx, pingA, N, cin, kpad, (int)Mrows);

        dim3 g64((unsigned)(Mrows / 64), 4);
        dim3 g32((unsigned)(Mrows / 64), 2);

        // Layer 1: (2*cin -> 64)
        zstats_kernel<<<1, 128, 0, stream>>>(ssum);
        if (kpad == 32)
            gemm_bn_kernel<32><<<g64, 128, 0, stream>>>(pingA, P(conv,0,0), P(conv,0,1),
                                                        pingB, ssum, ssq, cin2, 64, 64);
        else
            gemm_bn_kernel<64><<<g64, 128, 0, stream>>>(pingA, P(conv,0,0), P(conv,0,1),
                                                        pingB, ssum, ssq, cin2, 64, 64);
        stats_fin_kernel<<<1, 64, 0, stream>>>(ssum, ssq, scl, shf,
                                               P(conv,0,iGamma), P(conv,0,iBeta), 64, Mf);
        norm_kernel<<<(int)((Mrows * 64 + 255) / 256), 256, 0, stream>>>(
            pingB, scl, shf, (long)(Mrows * 64), 64);

        // Layer 2: (64 -> 64)
        zstats_kernel<<<1, 128, 0, stream>>>(ssum);
        gemm_bn_kernel<64><<<g64, 128, 0, stream>>>(pingB, P(conv,1,0), P(conv,1,1),
                                                    pingA, ssum, ssq, 64, 64, 64);
        stats_fin_kernel<<<1, 64, 0, stream>>>(ssum, ssq, scl, shf,
                                               P(conv,1,iGamma), P(conv,1,iBeta), 64, Mf);
        norm_kernel<<<(int)((Mrows * 64 + 255) / 256), 256, 0, stream>>>(
            pingA, scl, shf, (long)(Mrows * 64), 64);

        // Layer 3: (64 -> 32), normalization fused into max-over-K
        zstats_kernel<<<1, 128, 0, stream>>>(ssum);
        gemm_bn_kernel<64><<<g32, 128, 0, stream>>>(pingA, P(conv,2,0), P(conv,2,1),
                                                    pingB, ssum, ssq, 64, 32, 32);
        stats_fin_kernel<<<1, 64, 0, stream>>>(ssum, ssq, scl, shf,
                                               P(conv,2,iGamma), P(conv,2,iBeta), 32, Mf);
        maxk_kernel<<<(NB * NPTS * 32 + 255) / 256, 256, 0, stream>>>(
            pingB, scl, shf, hout, NB * NPTS * 32);
    };

    run_conv(pos, 3,  0, hA);
    run_conv(hA, 32,  1, hB);
    run_conv(hB, 32,  2, hA);

    gmax_kernel<<<NB * 32, 256, 0, stream>>>(hA, gbuf);
    linear_kernel<<<(NB * 512 + 255) / 256, 256, 0, stream>>>(gbuf, linW, linB, (float*)d_out);
}
